// GQAttention_86638080295420
// MI455X (gfx1250) — matrix operand order repaired and checked by reading, not compiled, not run on hardware
//
#include <hip/hip_runtime.h>
#include <hip/hip_bf16.h>

// ---------------------------------------------------------------------------
// GQA attention block for MI455X (gfx1250), bf16 WMMA path.
// B=2, S=2048, D=2048, H=16, KV=4, hd=128, rope theta=10000.
//
// Pipeline:
//   1) convert X -> bf16; transpose-convert [Wq|Wk|Wv] -> fused 3072xD bf16,
//      Wo -> Dx2048 bf16
//   2) one fused QKV GEMM (bf16 wmma, fp32 out, N=3072)
//   3) RoPE + repack (Q head-major, K head-major, V transposed per head)
//   4) flash attention: async double-buffered K/V tiles in LDS
//      (global_load_async_to_lds_b128 + s_wait_asynccnt), QK^T and PV on
//      v_wmma_f32_16x16x32_bf16, fp32 online softmax
//   5) output projection GEMM -> fp32 d_out
// ---------------------------------------------------------------------------

#define HD   128
#define NH   16
#define NKV  4
#define GQ   (NH / NKV)

typedef __attribute__((ext_vector_type(16))) __bf16 v16bf;
typedef __attribute__((ext_vector_type(8)))  __bf16 v8bf_t;
typedef __attribute__((ext_vector_type(8)))  float  v8f;

static __device__ __forceinline__ __bf16 f2bf(float f) {
    __hip_bfloat16 h = __float2bfloat16(f);
    return *reinterpret_cast<__bf16*>(&h);
}

// Async 16B global -> LDS copy (CDNA5 VGLOBAL async encoding, ASYNCcnt).
// ldsoff = LDS byte offset (low 32 bits of generic LDS address).
static __device__ __forceinline__ void async_copy16(const __bf16* g,
                                                    unsigned ldsoff) {
    asm volatile("global_load_async_to_lds_b128 %0, %1, off"
                 :: "v"(ldsoff), "v"(g)
                 : "memory");
}
static __device__ __forceinline__ void wait_async_le8() {
    asm volatile("s_wait_asynccnt 0x8" ::: "memory");
}
static __device__ __forceinline__ void wait_async_0() {
    asm volatile("s_wait_asynccnt 0x0" ::: "memory");
}

// ---------------------------------------------------------------------------
// Elementwise fp32 -> bf16
// ---------------------------------------------------------------------------
__global__ void cvt_f32_bf16(const float* __restrict__ in,
                             __bf16* __restrict__ out, long long n) {
    long long i = (long long)blockIdx.x * blockDim.x + threadIdx.x;
    if (i < n) out[i] = f2bf(in[i]);
}

// ---------------------------------------------------------------------------
// Transpose + convert: in is R x C fp32, out is C x R bf16 (out[c*R+r]=in[r*C+c]).
// ---------------------------------------------------------------------------
__global__ void transpose_cvt(const float* __restrict__ in,
                              __bf16* __restrict__ out, int R, int C) {
    long long i = (long long)blockIdx.x * blockDim.x + threadIdx.x;
    long long n = (long long)R * C;
    if (i >= n) return;
    int c = (int)(i / R);
    int r = (int)(i % R);
    out[i] = f2bf(in[(long long)r * C + c]);
}

// ---------------------------------------------------------------------------
// bf16 GEMM: C(MxN fp32) = A(MxK bf16 row-major) * Bt(NxK bf16).
// 128 threads = 4 waves (2x2); wave tile 32x64 = 2x4 wmma tiles.
// Block tile 64x128. M%64==0, N%128==0, K%32==0.
// ---------------------------------------------------------------------------
__global__ __launch_bounds__(128) void gemm_bf16(
    const __bf16* __restrict__ A, const __bf16* __restrict__ Bt,
    float* __restrict__ C, int M, int N, int K) {
    const int lane = threadIdx.x & 31;
    const int wave = threadIdx.x >> 5;
    const int half = lane >> 4;
    const int r    = lane & 15;
    const int m0   = blockIdx.y * 64 + (wave >> 1) * 32;
    const int n0   = blockIdx.x * 128 + (wave & 1) * 64;

    v8f acc[2][4] = {};

    for (int k0 = 0; k0 < K; k0 += 32) {
        if (k0 + 64 < K) {  // prefetch ahead (global_prefetch_b8)
            __builtin_prefetch(A + (size_t)(m0 + r) * K + k0 + 64, 0, 3);
            __builtin_prefetch(Bt + (size_t)(n0 + r) * K + k0 + 64, 0, 3);
        }
        v16bf a[2], b[4];
#pragma unroll
        for (int i = 0; i < 2; i++) {
            const __bf16* p = A + (size_t)(m0 + i * 16 + r) * K + k0 + half * 8;
            union { v16bf v; v8bf_t h[2]; } u;
            u.h[0] = *(const v8bf_t*)p;
            u.h[1] = *(const v8bf_t*)(p + 16);
            a[i] = u.v;
        }
#pragma unroll
        for (int j = 0; j < 4; j++) {
            const __bf16* p = Bt + (size_t)(n0 + j * 16 + r) * K + k0 + half * 8;
            union { v16bf v; v8bf_t h[2]; } u;
            u.h[0] = *(const v8bf_t*)p;
            u.h[1] = *(const v8bf_t*)(p + 16);
            b[j] = u.v;
        }
#pragma unroll
        for (int i = 0; i < 2; i++)
#pragma unroll
            for (int j = 0; j < 4; j++)
                acc[i][j] = __builtin_amdgcn_wmma_f32_16x16x32_bf16(
                    false, a[i], false, b[j], (short)0, acc[i][j], false, false);
    }

#pragma unroll
    for (int i = 0; i < 2; i++)
#pragma unroll
        for (int j = 0; j < 4; j++)
#pragma unroll
            for (int v = 0; v < 8; v++)
                C[(size_t)(m0 + i * 16 + v + 8 * half) * N + n0 + j * 16 + r] =
                    acc[i][j][v];
}

// ---------------------------------------------------------------------------
// RoPE + repack. QKVf is the fused projection output: (B*S) x 3072 fp32 with
// columns [0,2048)=Q, [2048,2560)=K, [2560,3072)=V.
// ---------------------------------------------------------------------------
#define LOG_THETA 9.210340371976184f   // ln(10000)
#define NQKV      (NH * HD + 2 * NKV * HD)   // 3072

__global__ void rope_q(const float* __restrict__ QKVf, __bf16* __restrict__ Qb,
                       int B, int S) {
    long long i = (long long)blockIdx.x * blockDim.x + threadIdx.x;
    long long n = (long long)B * NH * S * HD;
    if (i >= n) return;
    int d = (int)(i % HD);
    int s = (int)((i / HD) % S);
    int h = (int)((i / ((long long)HD * S)) % NH);
    int b = (int)(i / ((long long)HD * S * NH));
    const float* row = QKVf + ((size_t)b * S + s) * NQKV + h * HD;
    float x   = row[d];
    int   di  = d & 63;
    float inv = __expf(-(float)di * (LOG_THETA / 64.0f));
    float ang = (float)s * inv;
    float c, sn;
    __sincosf(ang, &sn, &c);
    float other = (d < 64) ? -row[d + 64] : row[d - 64];
    Qb[i] = f2bf(x * c + other * sn);
}

__global__ void rope_k(const float* __restrict__ QKVf, __bf16* __restrict__ Kb,
                       int B, int S) {
    long long i = (long long)blockIdx.x * blockDim.x + threadIdx.x;
    long long n = (long long)B * NKV * S * HD;
    if (i >= n) return;
    int d  = (int)(i % HD);
    int s  = (int)((i / HD) % S);
    int kv = (int)((i / ((long long)HD * S)) % NKV);
    int b  = (int)(i / ((long long)HD * S * NKV));
    const float* row = QKVf + ((size_t)b * S + s) * NQKV + NH * HD + kv * HD;
    float x   = row[d];
    int   di  = d & 63;
    float inv = __expf(-(float)di * (LOG_THETA / 64.0f));
    float ang = (float)s * inv;
    float c, sn;
    __sincosf(ang, &sn, &c);
    float other = (d < 64) ? -row[d + 64] : row[d - 64];
    Kb[i] = f2bf(x * c + other * sn);
}

// V columns of QKVf -> bf16 transposed per head: Vt[b][kv][d][s]
__global__ void vt_pack(const float* __restrict__ QKVf, __bf16* __restrict__ Vt,
                        int B, int S) {
    long long i = (long long)blockIdx.x * blockDim.x + threadIdx.x;
    long long n = (long long)B * NKV * HD * S;
    if (i >= n) return;
    int s  = (int)(i % S);
    int d  = (int)((i / S) % HD);
    int kv = (int)((i / ((long long)S * HD)) % NKV);
    int b  = (int)(i / ((long long)S * HD * NKV));
    Vt[i] = f2bf(QKVf[((size_t)b * S + s) * NQKV + NH * HD + NKV * HD +
                      kv * HD + d]);
}

// ---------------------------------------------------------------------------
// Flash attention: one (b, h, 64-row q tile) per block, 4 waves x 16 q rows.
// K/V tiles (32 keys) staged in LDS via async global->LDS DMA, double
// buffered; all 4 waves share each tile. QK^T: 8 wmma; PV: 8 wmma per iter.
// ---------------------------------------------------------------------------
__global__ __launch_bounds__(128) void flash_attn(
    const __bf16* __restrict__ Qb, const __bf16* __restrict__ Kb,
    const __bf16* __restrict__ Vt, const float* __restrict__ mask,
    __bf16* __restrict__ O, int S) {
    __shared__ __bf16 KtL[2][32 * 128];   // [key][dim]
    __shared__ __bf16 VtL[2][128 * 32];   // [dim][key]
    __shared__ __bf16 pbuf[4 * 16 * 32];  // per-wave P relayout buffer

    const int tid  = threadIdx.x;
    const int lane = tid & 31;
    const int wave = tid >> 5;
    const int half = lane >> 4;
    const int r    = lane & 15;
    const int qt   = blockIdx.x;
    const int h    = blockIdx.y;
    const int b    = blockIdx.z;
    const int kv   = h / GQ;
    const int qpos0 = qt * 64 + wave * 16;

    const __bf16* Qh = Qb + ((size_t)(b * NH + h) * S + qpos0) * HD;
    const __bf16* Kh = Kb + ((size_t)(b * NKV + kv) * S) * HD;
    const __bf16* Vh = Vt + ((size_t)(b * NKV + kv) * HD) * S;
    __bf16* pw = pbuf + wave * 16 * 32;

    // Issue async DMA for one 32-key K tile (contiguous 8KB) + V tile (strided)
    auto issue = [&](int kt, int buf) {
        const __bf16* gk = Kh + (size_t)kt * HD;
        unsigned lk = (unsigned)(uintptr_t)&KtL[buf][0];
        unsigned lv = (unsigned)(uintptr_t)&VtL[buf][0];
#pragma unroll
        for (int i = 0; i < 4; i++) {
            int cid = i * 128 + tid;                 // 512 x 16B chunks
            async_copy16(gk + cid * 8, lk + cid * 16);
            int vd = cid >> 2, vk = (cid & 3) * 8;   // V: 128 rows x 64B
            async_copy16(Vh + (size_t)vd * S + kt + vk, lv + cid * 16);
        }
    };

    // Q fragments: 4 K-chunks of 32 over hd=128, resident for whole kernel.
    v16bf qf[4];
#pragma unroll
    for (int c = 0; c < 4; c++) {
        const __bf16* p = Qh + (size_t)r * HD + c * 32 + half * 8;
        union { v16bf v; v8bf_t hh[2]; } u;
        u.hh[0] = *(const v8bf_t*)p;
        u.hh[1] = *(const v8bf_t*)(p + 16);
        qf[c] = u.v;
    }

    v8f o[8] = {};
    float rowmax[8], rowsum[8];
#pragma unroll
    for (int v = 0; v < 8; v++) { rowmax[v] = -3.0e38f; rowsum[v] = 0.0f; }

    const float scale = 0.08838834764831845f;  // 1/sqrt(128)

    issue(0, 0);
    int buf = 0;
    for (int kt = 0; kt < S; kt += 32) {
        const bool more = (kt + 32) < S;
        if (more) issue(kt + 32, buf ^ 1);
        if (more) wait_async_le8(); else wait_async_0();
        __syncthreads();

        // --- scores: two 16-key tiles, K fragments from LDS ---
        v8f sc[2] = {};
#pragma unroll
        for (int sub = 0; sub < 2; sub++) {
#pragma unroll
            for (int c = 0; c < 4; c++) {
                const __bf16* p =
                    &KtL[buf][(sub * 16 + r) * 128 + c * 32 + half * 8];
                union { v16bf v; v8bf_t hh[2]; } u;
                u.hh[0] = *(const v8bf_t*)p;
                u.hh[1] = *(const v8bf_t*)(p + 16);
                sc[sub] = __builtin_amdgcn_wmma_f32_16x16x32_bf16(
                    false, qf[c], false, u.v, (short)0, sc[sub], false, false);
            }
        }
        // --- scale + additive mask ---
#pragma unroll
        for (int v = 0; v < 8; v++) {
            int qrow = qpos0 + v + 8 * half;
            sc[0][v] = sc[0][v] * scale + mask[(size_t)qrow * S + kt + r];
            sc[1][v] = sc[1][v] * scale + mask[(size_t)qrow * S + kt + 16 + r];
        }
        // --- online softmax (row reductions across the 16-lane half) ---
        float al[8];
#pragma unroll
        for (int v = 0; v < 8; v++) {
            float m = fmaxf(sc[0][v], sc[1][v]);
#pragma unroll
            for (int off = 8; off >= 1; off >>= 1)
                m = fmaxf(m, __shfl_xor(m, off, 32));
            float mn = fmaxf(rowmax[v], m);
            al[v] = __expf(rowmax[v] - mn);
            rowmax[v] = mn;
            float p0 = __expf(sc[0][v] - mn);
            float p1 = __expf(sc[1][v] - mn);
            sc[0][v] = p0;
            sc[1][v] = p1;
            float rs = p0 + p1;
#pragma unroll
            for (int off = 8; off >= 1; off >>= 1)
                rs += __shfl_xor(rs, off, 32);
            rowsum[v] = rowsum[v] * al[v] + rs;
        }
#pragma unroll
        for (int nt = 0; nt < 8; nt++)
#pragma unroll
            for (int v = 0; v < 8; v++) o[nt][v] *= al[v];

        // --- P: C-layout regs -> LDS -> A-layout fragment (16x32) ---
#pragma unroll
        for (int v = 0; v < 8; v++) {
            pw[(v + 8 * half) * 32 + r]      = f2bf(sc[0][v]);
            pw[(v + 8 * half) * 32 + 16 + r] = f2bf(sc[1][v]);
        }
        union { v16bf v; v8bf_t hh[2]; } pu;
        pu.hh[0] = *(const v8bf_t*)&pw[r * 32 + half * 8];
        pu.hh[1] = *(const v8bf_t*)&pw[r * 32 + 16 + half * 8];
        v16bf pf = pu.v;

        // --- PV: V fragments from LDS, accumulate 8 n-tiles of 16x128 ---
#pragma unroll
        for (int nt = 0; nt < 8; nt++) {
            const __bf16* p = &VtL[buf][(nt * 16 + r) * 32 + half * 8];
            union { v16bf v; v8bf_t hh[2]; } u;
            u.hh[0] = *(const v8bf_t*)p;
            u.hh[1] = *(const v8bf_t*)(p + 16);
            o[nt] = __builtin_amdgcn_wmma_f32_16x16x32_bf16(
                false, pf, false, u.v, (short)0, o[nt], false, false);
        }
        __syncthreads();
        buf ^= 1;
    }

    // --- normalize and store bf16 in (B*S) x (NH*HD) token-major layout ---
#pragma unroll
    for (int v = 0; v < 8; v++) {
        float inv = 1.0f / rowsum[v];
        size_t row = (size_t)b * S + qpos0 + v + 8 * half;
#pragma unroll
        for (int nt = 0; nt < 8; nt++)
            O[row * (NH * HD) + h * HD + nt * 16 + r] = f2bf(o[nt][v] * inv);
    }
}

// ---------------------------------------------------------------------------
// Host-side launch
// ---------------------------------------------------------------------------
extern "C" void kernel_launch(void* const* d_in, const int* in_sizes, int n_in,
                              void* d_out, int out_size, void* d_ws,
                              size_t ws_size, hipStream_t stream) {
    const float* hs   = (const float*)d_in[0];
    const float* mask = (const float*)d_in[1];
    const float* Wq   = (const float*)d_in[2];
    const float* Wk   = (const float*)d_in[3];
    const float* Wv   = (const float*)d_in[4];
    const float* Wo   = (const float*)d_in[5];

    const int B = 2, S = 2048, D = 2048;
    const int M = B * S;          // 4096 token rows
    const int NQ = NH * HD;       // 2048
    const int NKVD = NKV * HD;    // 512
    const int NF = NQKV;          // 3072 fused QKV width

    char* wsp = (char*)d_ws;
    auto alloc = [&](size_t bytes) -> char* {
        char* p = wsp;
        wsp += (bytes + 255) & ~(size_t)255;
        return p;
    };
    __bf16* Xb    = (__bf16*)alloc((size_t)M * D * 2);
    __bf16* WqkvT = (__bf16*)alloc((size_t)NF * D * 2);   // rows: Q | K | V
    __bf16* WoT   = (__bf16*)alloc((size_t)D * NQ * 2);
    float*  QKVf  = (float*)alloc((size_t)M * NF * 4);
    __bf16* Qbf   = (__bf16*)alloc((size_t)M * NQ * 2);
    __bf16* Kbf   = (__bf16*)alloc((size_t)M * NKVD * 2);
    __bf16* Vtb   = (__bf16*)alloc((size_t)M * NKVD * 2);
    __bf16* Ob    = (__bf16*)alloc((size_t)M * NQ * 2);

    const int T = 256;
    // 1) conversions (fused QKV weight: rows [0,2048)=WqT, [2048,2560)=WkT,
    //    [2560,3072)=WvT — each block is an independent NxD transpose)
    {
        long long n = (long long)M * D;
        cvt_f32_bf16<<<(unsigned)((n + T - 1) / T), T, 0, stream>>>(hs, Xb, n);
        long long nw = (long long)D * NQ;
        transpose_cvt<<<(unsigned)((nw + T - 1) / T), T, 0, stream>>>(
            Wq, WqkvT, D, NQ);
        long long nk = (long long)D * NKVD;
        transpose_cvt<<<(unsigned)((nk + T - 1) / T), T, 0, stream>>>(
            Wk, WqkvT + (size_t)NQ * D, D, NKVD);
        transpose_cvt<<<(unsigned)((nk + T - 1) / T), T, 0, stream>>>(
            Wv, WqkvT + (size_t)(NQ + NKVD) * D, D, NKVD);
        transpose_cvt<<<(unsigned)((nw + T - 1) / T), T, 0, stream>>>(
            Wo, WoT, NQ, D);
    }
    // 2) fused QKV projection (bf16 WMMA, fp32 out, N=3072)
    {
        dim3 g(NF / 128, M / 64);
        gemm_bf16<<<g, 128, 0, stream>>>(Xb, WqkvT, QKVf, M, NF, D);
    }
    // 3) RoPE + repack to attention layouts
    {
        long long nq = (long long)B * NH * S * HD;
        rope_q<<<(unsigned)((nq + T - 1) / T), T, 0, stream>>>(QKVf, Qbf, B, S);
        long long nk = (long long)B * NKV * S * HD;
        rope_k<<<(unsigned)((nk + T - 1) / T), T, 0, stream>>>(QKVf, Kbf, B, S);
        vt_pack<<<(unsigned)((nk + T - 1) / T), T, 0, stream>>>(QKVf, Vtb, B, S);
    }
    // 4) flash attention (async double-buffered K/V tiles in LDS)
    {
        dim3 g(S / 64, NH, B);
        flash_attn<<<g, 128, 0, stream>>>(Qbf, Kbf, Vtb, mask, Ob, S);
    }
    // 5) output projection -> d_out (fp32)
    {
        dim3 g(D / 128, M / 64);
        gemm_bf16<<<g, 128, 0, stream>>>(Ob, WoT, (float*)d_out, M, D, NQ);
    }
}
